// FlashFFN_23699629539601
// MI455X (gfx1250) — matrix operand order repaired and checked by reading, not compiled, not run on hardware
//
#include <hip/hip_runtime.h>
#include <hip/hip_bf16.h>

// ---------------------------------------------------------------------------
// FlashFFN for MI455X (gfx1250, wave32, WMMA + async global->LDS copies).
//   out = gelu(x @ W1 + b1) @ W2 + b2
//   HIDDEN=4096, FFN=16384, T = B*S = 4096 tokens.
// bf16 WMMA (v_wmma_f32_16x16x32_bf16), fp32 accumulation, double-buffered
// LDS staging via global_load_async_to_lds_b128 (ASYNCcnt).
// ---------------------------------------------------------------------------

#define HID   4096
#define FFN_D 16384
#define TOK   4096

typedef __attribute__((ext_vector_type(16))) __bf16 v16bf;
typedef __attribute__((ext_vector_type(8)))  __bf16 bf16x8;
typedef __attribute__((ext_vector_type(4)))  __bf16 bf16x4;
typedef __attribute__((ext_vector_type(8)))  float  v8f;

// Async 16-byte global->LDS copy (per active lane). VDST = LDS byte address
// (low 32 bits of the generic pointer == LDS offset per ISA §10.2).
__device__ __forceinline__ void async_b128(unsigned lds_off, const void* gaddr) {
    asm volatile("global_load_async_to_lds_b128 %0, %1, off"
                 :: "v"(lds_off), "v"(gaddr) : "memory");
}
__device__ __forceinline__ void wait_async0() {
#if __has_builtin(__builtin_amdgcn_s_wait_asynccnt)
    __builtin_amdgcn_s_wait_asynccnt(0);
#else
    asm volatile("s_wait_asynccnt 0" ::: "memory");
#endif
}

// ---------------------------------------------------------------------------
// fp32 -> bf16 elementwise (row-major preserved). 4 elements / thread.
// ---------------------------------------------------------------------------
__global__ __launch_bounds__(256) void cvt_f32_to_bf16(const float* __restrict__ in,
                                                       __bf16* __restrict__ out,
                                                       int n4) {
    int i = blockIdx.x * 256 + threadIdx.x;
    if (i >= n4) return;
    const float4 v = ((const float4*)in)[i];
    bf16x4 o;
    o.x = (__bf16)v.x; o.y = (__bf16)v.y; o.z = (__bf16)v.z; o.w = (__bf16)v.w;
    ((bf16x4*)out)[i] = o;
}

// ---------------------------------------------------------------------------
// fp32 [R][C] -> bf16 [C][R] (transpose + convert), 64x64 LDS tiles.
// Produces the [N][K] operand so GEMM fragment reads are K-contiguous.
// ---------------------------------------------------------------------------
__global__ __launch_bounds__(256) void transpose_cvt(const float* __restrict__ in,
                                                     __bf16* __restrict__ out,
                                                     int R, int C) {
    __shared__ __bf16 tile[64 * 65];
    const int r0 = blockIdx.y * 64;
    const int c0 = blockIdx.x * 64;
    #pragma unroll
    for (int p = 0; p < 16; ++p) {
        int li = p * 256 + threadIdx.x;
        int r = li >> 6, c = li & 63;
        tile[r * 65 + c] = (__bf16)in[(size_t)(r0 + r) * C + (c0 + c)];
    }
    __syncthreads();
    #pragma unroll
    for (int p = 0; p < 16; ++p) {
        int li = p * 256 + threadIdx.x;
        int c = li >> 6, r = li & 63;
        out[(size_t)(c0 + c) * R + (r0 + r)] = tile[r * 65 + c];
    }
}

// ---------------------------------------------------------------------------
// Tiled bf16 WMMA GEMM:  C[M][N] = act(A[M][K] @ Bt[N][K]^T + bias[N])
// Block: 256 threads (8 waves), 256(M) x 128(N) tile, K-step 32.
// Wave grid 4(M) x 2(N); each wave owns 64x64 = 4x4 fragments of 16x16
// -> 16 v_wmma per K-step against 16 ds_load_b128.
// Double-buffered LDS, filled by async global->LDS copies.
// ---------------------------------------------------------------------------
template <bool GELU, typename OutT>
__global__ __launch_bounds__(256) void ffn_gemm(const __bf16* __restrict__ A,
                                                const __bf16* __restrict__ Bt,
                                                const float*  __restrict__ bias,
                                                OutT* __restrict__ C,
                                                int M, int N, int K) {
    constexpr int TK = 32;                 // K-step
    constexpr int LD = 40;                 // 32 + 8 pad; rows 80B apart (20 banks)
    constexpr int ASZ = 256 * LD;          // one A buffer, elements
    constexpr int BSZ = 128 * LD;          // one B buffer, elements
    __shared__ __bf16 As[2 * ASZ];         // 40 KB
    __shared__ __bf16 Bs[2 * BSZ];         // 20 KB

    const int tid   = threadIdx.x;
    const int wave  = tid >> 5;
    const int lane  = tid & 31;
    const int waveM = wave >> 1;           // 0..3 -> 64-row slice
    const int waveN = wave & 1;            // 0..1 -> 64-col slice
    const int half  = lane >> 4;           // ISA lane-half for WMMA layouts
    const int l16   = lane & 15;

    const int mBase = blockIdx.y * 256;
    const int nBase = blockIdx.x * 128;

    // staging: 4 threads per 32-wide row, 8 bf16 (16B) each
    const int rowL = tid >> 2;             // 0..63
    const int colL = (tid & 3) * 8;        // 0,8,16,24

    const unsigned asBase = (unsigned)(size_t)&As[0];
    const unsigned bsBase = (unsigned)(size_t)&Bs[0];

    // Issue all async copies for K-tile k0 into buffer `buf` (no waiting).
    auto stage = [&](int k0, int buf) {
        #pragma unroll
        for (int p = 0; p < 4; ++p) {      // A: 256 rows x 32 cols
            int r = rowL + p * 64;
            async_b128(asBase + (unsigned)(buf * ASZ + r * LD + colL) * 2,
                       &A[(size_t)(mBase + r) * K + k0 + colL]);
        }
        #pragma unroll
        for (int p = 0; p < 2; ++p) {      // B: 128 rows x 32 cols
            int r = rowL + p * 64;
            async_b128(bsBase + (unsigned)(buf * BSZ + r * LD + colL) * 2,
                       &Bt[(size_t)(nBase + r) * K + k0 + colL]);
        }
    };

    v8f acc[4][4];
    #pragma unroll
    for (int i = 0; i < 4; ++i)
        #pragma unroll
        for (int j = 0; j < 4; ++j)
            acc[i][j] = (v8f){0.f, 0.f, 0.f, 0.f, 0.f, 0.f, 0.f, 0.f};

    const int NT = K / TK;
    stage(0, 0);                           // prologue: tile 0 in flight

    for (int t = 0; t < NT; ++t) {
        const int buf = t & 1;
        wait_async0();                     // own slice of tile t landed in LDS
        __syncthreads();                   // everyone's slice landed; prev reads done
        if (t + 1 < NT) stage((t + 1) * TK, buf ^ 1);  // overlap with compute

        // A fragment (16x32): e<8 -> K=half*8+e ; e>=8 -> K=16+half*8+(e-8)
        v16bf af[4];
        #pragma unroll
        for (int i = 0; i < 4; ++i) {
            int r = waveM * 64 + i * 16 + l16;
            const int idx = buf * ASZ + r * LD + half * 8;
            bf16x8 lo = *(const bf16x8*)&As[idx];
            bf16x8 hi = *(const bf16x8*)&As[idx + 16];
            af[i] = __builtin_shufflevector(lo, hi, 0, 1, 2, 3, 4, 5, 6, 7,
                                            8, 9, 10, 11, 12, 13, 14, 15);
        }
        // B fragment (32x16): lane column n=l16, e<8 -> K=half*8+e ; e>=8 -> K=16+half*8+(e-8)
        v16bf bf[4];
        #pragma unroll
        for (int j = 0; j < 4; ++j) {
            int n = waveN * 64 + j * 16 + l16;
            const int idx = buf * BSZ + n * LD + half * 8;
            bf16x8 lo = *(const bf16x8*)&Bs[idx];
            bf16x8 hi = *(const bf16x8*)&Bs[idx + 16];
            bf[j] = __builtin_shufflevector(lo, hi, 0, 1, 2, 3, 4, 5, 6, 7,
                                            8, 9, 10, 11, 12, 13, 14, 15);
        }

        #pragma unroll
        for (int i = 0; i < 4; ++i)
            #pragma unroll
            for (int j = 0; j < 4; ++j)
                acc[i][j] = __builtin_amdgcn_wmma_f32_16x16x32_bf16(
                    false, af[i], false, bf[j], (short)0, acc[i][j], false, false);
    }

    // Epilogue: C/D layout -> lane l16 = column, VGPR r -> row r + 8*half.
    #pragma unroll
    for (int j = 0; j < 4; ++j) {
        int n = nBase + waveN * 64 + j * 16 + l16;
        float bv = bias[n];
        #pragma unroll
        for (int i = 0; i < 4; ++i) {
            #pragma unroll
            for (int r = 0; r < 8; ++r) {
                int m = mBase + waveM * 64 + i * 16 + half * 8 + r;
                float v = acc[i][j][r] + bv;
                if (GELU)
                    v = 0.5f * v * (1.0f + erff(v * 0.70710678118654752f));
                C[(size_t)m * N + n] = (OutT)v;
            }
        }
    }
}

// ---------------------------------------------------------------------------
// Launch: inputs x, W1, b1, W2, b2 (fp32). Output fp32 [T][HID].
// Workspace (bf16): Xb 32MB | W1t 128MB | W2t 128MB | Hb 128MB.
// ---------------------------------------------------------------------------
extern "C" void kernel_launch(void* const* d_in, const int* in_sizes, int n_in,
                              void* d_out, int out_size, void* d_ws, size_t ws_size,
                              hipStream_t stream) {
    const float* x  = (const float*)d_in[0];
    const float* W1 = (const float*)d_in[1];
    const float* b1 = (const float*)d_in[2];
    const float* W2 = (const float*)d_in[3];
    const float* b2 = (const float*)d_in[4];
    float* out = (float*)d_out;

    char* ws = (char*)d_ws;
    __bf16* Xb  = (__bf16*)(ws);
    __bf16* W1t = (__bf16*)(ws + (size_t)TOK * HID * 2);
    __bf16* W2t = (__bf16*)(ws + (size_t)TOK * HID * 2 + (size_t)HID * FFN_D * 2);
    __bf16* Hb  = (__bf16*)(ws + (size_t)TOK * HID * 2 + 2 * (size_t)HID * FFN_D * 2);

    // 1) x -> bf16 (row-major [T][HID])
    {
        int n4 = (TOK * HID) / 4;
        cvt_f32_to_bf16<<<n4 / 256, 256, 0, stream>>>(x, Xb, n4);
    }
    // 2) W1 [HID][FFN] -> W1t [FFN][HID] bf16 ; W2 [FFN][HID] -> W2t [HID][FFN] bf16
    transpose_cvt<<<dim3(FFN_D / 64, HID / 64), 256, 0, stream>>>(W1, W1t, HID, FFN_D);
    transpose_cvt<<<dim3(HID / 64, FFN_D / 64), 256, 0, stream>>>(W2, W2t, FFN_D, HID);

    // 3) H = gelu(Xb @ W1 + b1)  -> bf16 [T][FFN]
    ffn_gemm<true, __bf16><<<dim3(FFN_D / 128, TOK / 256), 256, 0, stream>>>(
        Xb, W1t, b1, Hb, TOK, FFN_D, HID);

    // 4) out = Hb @ W2 + b2      -> fp32 [T][HID]
    ffn_gemm<false, float><<<dim3(HID / 128, TOK / 256), 256, 0, stream>>>(
        Hb, W2t, b2, out, TOK, HID, FFN_D);

    (void)in_sizes; (void)n_in; (void)out_size; (void)ws_size;
}
